// MixtureOfExperts_23957327577822
// MI455X (gfx1250) — matrix operand order repaired and checked by reading, not compiled, not run on hardware
//
#include <hip/hip_runtime.h>
#include <hip/hip_bf16.h>

// Problem constants (from reference): B=4, S=2048, D=1024, E=8, K=2, H=1408
#define T_TOK 8192
#define DDIM  1024
#define EEXP  8
#define HDIM  1408
#define HCHUNK 704          // H processed in 2 chunks so bf16 activations fit LDS
#define MAXTILES 1032       // 2*T/16 + E (worst-case expert tiles, top-2 routing)

typedef __attribute__((ext_vector_type(16))) __bf16 v16bf;
typedef __attribute__((ext_vector_type(8)))  float  v8f;

union V16 { uint4 q[2]; v16bf v; };

__device__ __forceinline__ unsigned short f2bf(float f) {
  unsigned int u = __float_as_uint(f);
  u += 0x7FFFu + ((u >> 16) & 1u);   // round-to-nearest-even
  return (unsigned short)(u >> 16);
}

__device__ __forceinline__ float silu(float v) {
  return v / (1.0f + __expf(-v));
}

// meta layout (32 ints): [0..7]=count  [8..15]=cursor  [16..23]=base slot  [24]=ntiles
// ---------------------------------------------------------------------------
// Zero output (exactly T*D floats) and routing metadata.
// ---------------------------------------------------------------------------
__global__ void __launch_bounds__(256)
moe_zero_kernel(float* __restrict__ out, int* __restrict__ meta) {
  const size_t i = ((size_t)blockIdx.x * 256 + threadIdx.x) * 4;
  float4 z = {0.0f, 0.0f, 0.0f, 0.0f};
  *(float4*)(out + i) = z;
  if (blockIdx.x == 0 && threadIdx.x < 32) meta[threadIdx.x] = 0;
}

// ---------------------------------------------------------------------------
// Router: probs = softmax(x @ Wr), top-2 renormalized combine weights.
// ---------------------------------------------------------------------------
__global__ void __launch_bounds__(256)
moe_router_kernel(const float* __restrict__ x, const float* __restrict__ Wr,
                  float* __restrict__ combine, float* __restrict__ probs) {
  __shared__ float wrs[DDIM * EEXP];     // 32 KB: whole router weight in LDS
  const int tid = threadIdx.x;
  for (int i = tid; i < DDIM * EEXP; i += 256) wrs[i] = Wr[i];
  __syncthreads();

  const int t = blockIdx.x * 256 + tid;  // exact grid: T_TOK/256 blocks
  float acc[EEXP];
#pragma unroll
  for (int e = 0; e < EEXP; ++e) acc[e] = 0.0f;

  const float* xr = x + (size_t)t * DDIM;
  for (int d = 0; d < DDIM; ++d) {
    const float xv = xr[d];
#pragma unroll
    for (int e = 0; e < EEXP; ++e) acc[e] = fmaf(xv, wrs[d * EEXP + e], acc[e]);
  }

  float mx = acc[0];
#pragma unroll
  for (int e = 1; e < EEXP; ++e) mx = fmaxf(mx, acc[e]);
  float p[EEXP], sum = 0.0f;
#pragma unroll
  for (int e = 0; e < EEXP; ++e) { p[e] = __expf(acc[e] - mx); sum += p[e]; }
  const float inv = 1.0f / sum;
#pragma unroll
  for (int e = 0; e < EEXP; ++e) { p[e] *= inv; probs[(size_t)t * EEXP + e] = p[e]; }

  int i1 = 0; float p1 = p[0];
#pragma unroll
  for (int e = 1; e < EEXP; ++e) if (p[e] > p1) { p1 = p[e]; i1 = e; }
  int i2 = (i1 == 0) ? 1 : 0; float p2 = p[i2];
#pragma unroll
  for (int e = 0; e < EEXP; ++e) if (e != i1 && p[e] > p2) { p2 = p[e]; i2 = e; }
  const float nrm = 1.0f / (p1 + p2);
#pragma unroll
  for (int e = 0; e < EEXP; ++e) combine[(size_t)t * EEXP + e] = 0.0f;
  combine[(size_t)t * EEXP + i1] = p1 * nrm;
  combine[(size_t)t * EEXP + i2] = p2 * nrm;
}

// ---------------------------------------------------------------------------
// Load-balance loss: deterministic single-block tree reduction over probs.
// ---------------------------------------------------------------------------
__global__ void __launch_bounds__(256)
moe_lb_kernel(const float* __restrict__ probs, float* __restrict__ lb_out) {
  __shared__ float red[256][EEXP];
  const int tid = threadIdx.x;
  float u[EEXP];
#pragma unroll
  for (int e = 0; e < EEXP; ++e) u[e] = 0.0f;
  for (int r = tid; r < T_TOK; r += 256) {
#pragma unroll
    for (int e = 0; e < EEXP; ++e) u[e] += probs[(size_t)r * EEXP + e];
  }
#pragma unroll
  for (int e = 0; e < EEXP; ++e) red[tid][e] = u[e];
  __syncthreads();
  for (int s = 128; s > 0; s >>= 1) {
    if (tid < s) {
#pragma unroll
      for (int e = 0; e < EEXP; ++e) red[tid][e] += red[tid + s][e];
    }
    __syncthreads();
  }
  if (tid == 0) {
    float lb = 0.0f;
#pragma unroll
    for (int e = 0; e < EEXP; ++e) {
      const float d = red[0][e] / (float)T_TOK - 1.0f / (float)EEXP;
      lb += d * d;
    }
    lb_out[0] = 0.01f * (lb / (float)EEXP);
  }
}

// ---------------------------------------------------------------------------
// Per-expert token counting (integer atomics -> deterministic counts).
// ---------------------------------------------------------------------------
__global__ void __launch_bounds__(256)
moe_count_kernel(const float* __restrict__ combine, int* __restrict__ meta) {
  const int t = blockIdx.x * 256 + threadIdx.x;
#pragma unroll
  for (int e = 0; e < EEXP; ++e)
    if (combine[(size_t)t * EEXP + e] != 0.0f) atomicAdd(&meta[e], 1);
}

// ---------------------------------------------------------------------------
// Plan: bases, tile table (expert id + start slot), pad tail slots with -1.
// ---------------------------------------------------------------------------
__global__ void moe_plan_kernel(int* __restrict__ meta, int* __restrict__ tile_e,
                                int* __restrict__ tile_s, int* __restrict__ idx_list) {
  if (threadIdx.x != 0) return;
  int b = 0, nt = 0;
  for (int e = 0; e < EEXP; ++e) {
    const int c = meta[e];
    meta[16 + e] = b;                       // base slot for expert e
    const int tiles = (c + 15) >> 4;
    for (int ti = 0; ti < tiles; ++ti) { tile_e[nt] = e; tile_s[nt] = b + ti * 16; ++nt; }
    for (int s = c; s < tiles * 16; ++s) idx_list[b + s] = -1;   // pads
    b += tiles * 16;
  }
  meta[24] = nt;
}

// ---------------------------------------------------------------------------
// Scatter tokens into per-expert slot lists. Slot order is nondeterministic
// but the per-(token,expert) contribution is slot-invariant, and each output
// element receives exactly two commutative float adds -> output deterministic.
// ---------------------------------------------------------------------------
__global__ void __launch_bounds__(256)
moe_scatter_kernel(const float* __restrict__ combine, int* __restrict__ meta,
                   int* __restrict__ idx_list) {
  const int t = blockIdx.x * 256 + threadIdx.x;
#pragma unroll
  for (int e = 0; e < EEXP; ++e)
    if (combine[(size_t)t * EEXP + e] != 0.0f) {
      const int p = atomicAdd(&meta[8 + e], 1);
      idx_list[meta[16 + e] + p] = t;
    }
}

// ---------------------------------------------------------------------------
// Cast x -> bf16 (row-major [T, D]).
// ---------------------------------------------------------------------------
__global__ void __launch_bounds__(256)
moe_cast_x_kernel(const float* __restrict__ x, unsigned short* __restrict__ xb) {
  const size_t b = ((size_t)blockIdx.x * 256 + threadIdx.x) * 8;
  const float* p = x + b;
  uint4 o;
  o.x = (unsigned)f2bf(p[0]) | ((unsigned)f2bf(p[1]) << 16);
  o.y = (unsigned)f2bf(p[2]) | ((unsigned)f2bf(p[3]) << 16);
  o.z = (unsigned)f2bf(p[4]) | ((unsigned)f2bf(p[5]) << 16);
  o.w = (unsigned)f2bf(p[6]) | ((unsigned)f2bf(p[7]) << 16);
  *(uint4*)(xb + b) = o;
}

// ---------------------------------------------------------------------------
// Cast + transpose weights: fp32 [E, R, C] -> bf16 [E, C, R] (coalesced stores).
// ---------------------------------------------------------------------------
__global__ void __launch_bounds__(256)
moe_cast_transpose_kernel(const float* __restrict__ in, unsigned short* __restrict__ outp,
                          int R, int C) {
  const int o = blockIdx.x * 256 + threadIdx.x;   // grid is exact
  const int r = o % R;
  const int t = o / R;
  const int c = t % C;
  const int e = t / C;
  outp[o] = f2bf(in[((size_t)e * R + r) * C + c]);
}

// ---------------------------------------------------------------------------
// Gathered expert kernel. One block = one (expert, 16-token) tile; 8 waves.
//   x rows gathered into LDS via GLOBAL_LOAD_ASYNC_TO_LDS_B128 (ASYNCcnt).
//   GEMM1: [16,D]@[D,Hchunk] x3 (gate/up/expand, WMMA bf16) -> SiLU mix,
//          scaled by combine[t,e] & alpha[e], bf16 into LDS.
//   GEMM2: [16,Hchunk]@[Hchunk,D] accumulated in v8f regs over both chunks.
//   Output scattered with global_atomic_add_f32 (exactly 2 adds per element).
// Weights pre-transposed to [N,K] bf16 so each lane's 16 K-values are
// contiguous (2x global_load_b128 per WMMA B operand).
// ---------------------------------------------------------------------------
__global__ void __launch_bounds__(256)
moe_expert_kernel(const unsigned short* __restrict__ xb,
                  const unsigned short* __restrict__ Wgt,
                  const unsigned short* __restrict__ Wut,
                  const unsigned short* __restrict__ Wxt,
                  const unsigned short* __restrict__ Wdt,
                  const float* __restrict__ alpha,
                  const float* __restrict__ combine,
                  const int* __restrict__ meta,
                  const int* __restrict__ tile_e,
                  const int* __restrict__ tile_s,
                  const int* __restrict__ idx_list,
                  float* __restrict__ out) {
  __shared__ unsigned short xs[16 * DDIM];     // 32 KB gathered x tile (bf16)
  __shared__ unsigned short comb[16 * HCHUNK]; // 22 KB combined activations (bf16)
  __shared__ int   tok[16];
  __shared__ float cws[16];

  if (blockIdx.x >= meta[24]) return;          // uniform early-exit past ntiles

  const int tid  = threadIdx.x;
  const int wave = tid >> 5;
  const int lane = tid & 31;
  const int nloc = lane & 15;      // B-column / A-row for this lane
  const int hi   = lane >> 4;      // half-wave selector (contiguous K: k = 16*hi + s)

  const int e  = tile_e[blockIdx.x];
  const int s0 = tile_s[blockIdx.x];
  if (tid < 16) {
    const int t = idx_list[s0 + tid];
    tok[tid] = (t < 0) ? 0 : t;                              // pad -> token 0
    cws[tid] = (t < 0) ? 0.0f : combine[(size_t)t * EEXP + e]; // pad weight 0
  }
  __syncthreads();

  // Gather 16 token rows into LDS with async copy (16 B per lane per issue).
  // Low 32 bits of a generic shared-aperture pointer are the LDS address.
  for (int i = tid; i < (16 * DDIM) / 8; i += 256) {
    const int r = i >> 7, c = i & 127;                       // 128 x 16B per row
    const unsigned long long ga =
        (unsigned long long)(xb + (size_t)tok[r] * DDIM + c * 8);
    const unsigned ldsa = (unsigned)(size_t)(void*)&xs[r * DDIM + c * 8];
    asm volatile("global_load_async_to_lds_b128 %0, %1, off"
                 :: "v"(ldsa), "v"(ga) : "memory");
  }
  asm volatile("s_wait_asynccnt 0x0" ::: "memory");
  __syncthreads();

  const float ae = alpha[e];
  v8f zero = {};
  v8f oacc[8];
#pragma unroll
  for (int j = 0; j < 8; ++j) oacc[j] = zero;

  for (int c2 = 0; c2 < 2; ++c2) {
    // ---------------- GEMM1 + activation epilogue ----------------
    for (int hn = wave; hn < HCHUNK / 16; hn += 8) {
      const int ncol = c2 * HCHUNK + hn * 16 + nloc;   // global H column
      const unsigned short* pg = Wgt + ((size_t)e * HDIM + ncol) * DDIM + hi * 16;
      const unsigned short* pu = Wut + ((size_t)e * HDIM + ncol) * DDIM + hi * 16;
      const unsigned short* px = Wxt + ((size_t)e * HDIM + ncol) * DDIM + hi * 16;
      v8f ag = zero, au = zero, ax = zero;
      for (int k0 = 0; k0 < DDIM; k0 += 32) {
        V16 a;
        const unsigned short* ap = xs + nloc * DDIM + k0 + hi * 16;
        a.q[0] = *(const uint4*)ap;
        a.q[1] = *(const uint4*)(ap + 8);
        v16bf bg = *(const v16bf*)(pg + k0);
        v16bf bu = *(const v16bf*)(pu + k0);
        v16bf bx = *(const v16bf*)(px + k0);
        __builtin_prefetch((const void*)(pg + k0 + 64), 0, 1);
        __builtin_prefetch((const void*)(pu + k0 + 64), 0, 1);
        ag = __builtin_amdgcn_wmma_f32_16x16x32_bf16(false, a.v, false, bg, (short)0, ag, false, false);
        au = __builtin_amdgcn_wmma_f32_16x16x32_bf16(false, a.v, false, bu, (short)0, au, false, false);
        ax = __builtin_amdgcn_wmma_f32_16x16x32_bf16(false, a.v, false, bx, (short)0, ax, false, false);
      }
      // combined = combine[t,e] * up * (alpha*silu(gate) + (1-alpha)*silu(expand))
#pragma unroll
      for (int i = 0; i < 8; ++i) {
        const int m = i + hi * 8;                       // C-layout row
        const float g = ag[i], u = au[i], xv = ax[i];
        const float val = cws[m] * (u * (ae * silu(g) + (1.0f - ae) * silu(xv)));
        comb[m * HCHUNK + hn * 16 + nloc] = f2bf(val);
      }
    }
    __syncthreads();

    // ---------------- GEMM2: out-tile += combined @ Wd ----------------
#pragma unroll
    for (int j = 0; j < 8; ++j) {
      const int ncol = (wave * 8 + j) * 16 + nloc;      // output D column
      const unsigned short* pd =
          Wdt + ((size_t)e * DDIM + ncol) * HDIM + c2 * HCHUNK + hi * 16;
      for (int k0 = 0; k0 < HCHUNK; k0 += 32) {
        V16 a;
        const unsigned short* ap = comb + nloc * HCHUNK + k0 + hi * 16;
        a.q[0] = *(const uint4*)ap;
        a.q[1] = *(const uint4*)(ap + 8);
        v16bf bd = *(const v16bf*)(pd + k0);
        __builtin_prefetch((const void*)(pd + k0 + 64), 0, 1);
        oacc[j] = __builtin_amdgcn_wmma_f32_16x16x32_bf16(false, a.v, false, bd, (short)0, oacc[j], false, false);
      }
    }
    __syncthreads();   // comb reused next chunk
  }

  // Scatter: exactly 2 expert contributions per output element (pads add +0.0).
#pragma unroll
  for (int j = 0; j < 8; ++j) {
    const int ncol = (wave * 8 + j) * 16 + nloc;
#pragma unroll
    for (int i = 0; i < 8; ++i) {
      const int m = i + hi * 8;
      atomicAdd(&out[(size_t)tok[m] * DDIM + ncol], oacc[j][i]);
    }
  }
}

// ---------------------------------------------------------------------------
extern "C" void kernel_launch(void* const* d_in, const int* in_sizes, int n_in,
                              void* d_out, int out_size, void* d_ws, size_t ws_size,
                              hipStream_t stream) {
  (void)in_sizes; (void)n_in; (void)out_size; (void)ws_size;
  const float* x     = (const float*)d_in[0];
  const float* Wr    = (const float*)d_in[1];
  const float* Wg    = (const float*)d_in[2];
  const float* Wu    = (const float*)d_in[3];
  const float* Wx    = (const float*)d_in[4];
  const float* Wd    = (const float*)d_in[5];
  const float* alpha = (const float*)d_in[6];
  float* out = (float*)d_out;

  // Workspace (~109 MB): routing data + bf16 x + 4 bf16 transposed weights.
  char* ws = (char*)d_ws;
  size_t off = 0;
  auto bump = [&](size_t bytes) { size_t c = off; off = (off + bytes + 255) & ~(size_t)255; return c; };
  float* combine = (float*)(ws + bump((size_t)T_TOK * EEXP * 4));
  float* probs   = (float*)(ws + bump((size_t)T_TOK * EEXP * 4));
  unsigned short* xb = (unsigned short*)(ws + bump((size_t)T_TOK * DDIM * 2));
  const size_t wbytes = (size_t)EEXP * DDIM * HDIM * 2;
  unsigned short* wgt = (unsigned short*)(ws + bump(wbytes));
  unsigned short* wut = (unsigned short*)(ws + bump(wbytes));
  unsigned short* wxt = (unsigned short*)(ws + bump(wbytes));
  unsigned short* wdt = (unsigned short*)(ws + bump(wbytes));
  int* meta     = (int*)(ws + bump(32 * 4));
  int* tile_e   = (int*)(ws + bump(MAXTILES * 4));
  int* tile_s   = (int*)(ws + bump(MAXTILES * 4));
  int* idx_list = (int*)(ws + bump((size_t)MAXTILES * 16 * 4));

  moe_zero_kernel<<<(T_TOK * DDIM / 4) / 256, 256, 0, stream>>>(out, meta);
  moe_router_kernel<<<T_TOK / 256, 256, 0, stream>>>(x, Wr, combine, probs);
  moe_lb_kernel<<<1, 256, 0, stream>>>(probs, out + (size_t)T_TOK * DDIM);
  moe_count_kernel<<<T_TOK / 256, 256, 0, stream>>>(combine, meta);
  moe_plan_kernel<<<1, 1, 0, stream>>>(meta, tile_e, tile_s, idx_list);
  moe_scatter_kernel<<<T_TOK / 256, 256, 0, stream>>>(combine, meta, idx_list);
  moe_cast_x_kernel<<<(T_TOK * DDIM / 8) / 256, 256, 0, stream>>>(x, xb);

  const int tblocks = (EEXP * DDIM * HDIM) / 256;
  moe_cast_transpose_kernel<<<tblocks, 256, 0, stream>>>(Wg, wgt, DDIM, HDIM);
  moe_cast_transpose_kernel<<<tblocks, 256, 0, stream>>>(Wu, wut, DDIM, HDIM);
  moe_cast_transpose_kernel<<<tblocks, 256, 0, stream>>>(Wx, wxt, DDIM, HDIM);
  moe_cast_transpose_kernel<<<tblocks, 256, 0, stream>>>(Wd, wdt, HDIM, DDIM);

  moe_expert_kernel<<<MAXTILES, 256, 0, stream>>>(xb, wgt, wut, wxt, wdt, alpha,
                                                  combine, meta, tile_e, tile_s,
                                                  idx_list, out);
}
